// Qwen3MoeDecoderLayer_32916629356774
// MI455X (gfx1250) — compile-verified, matrix operand order checked by reading, not run on hardware
//
#include <hip/hip_runtime.h>
#include <hip/hip_bf16.h>

// ---------------------------------------------------------------------------
// Qwen3-MoE decoder layer for MI455X (gfx1250, wave32, WMMA bf16 16x16x32).
// Layer is HBM-bound on MoE weights (~1.2 GB fp32): every weight is read
// exactly once. B tiles are staged to LDS with CDNA5 async global->LDS
// (GLOBAL_LOAD_ASYNC_TO_LDS_B128, ASYNCcnt), double-buffered so the async
// engine streams the next tile while WMMAs consume the current one.
// fp32->bf16 conversion happens on fragment read from LDS.
// ---------------------------------------------------------------------------

typedef __bf16 bf16_t;
typedef __attribute__((ext_vector_type(16))) __bf16 v16bf;
typedef __attribute__((ext_vector_type(8)))  __bf16 v8bf;
typedef __attribute__((ext_vector_type(8)))  float  v8f;

// Exact pointee type expected by the async-to-LDS builtin (from hipcc
// diagnostic: "__attribute__((__vector_size__(4 * sizeof(int)))) int").
typedef int v4i32 __attribute__((vector_size(16)));
typedef __attribute__((address_space(1))) v4i32* as1_v4i32p;  // global
typedef __attribute__((address_space(3))) v4i32* as3_v4i32p;  // LDS

constexpr int cB = 2, cS = 1024, cD = 2048;
constexpr int cH = 16, cKV = 4, cHD = 128;
constexpr int cE = 64, cK = 8, cI = 768;
constexpr int cT = cB * cS;                 // 2048 tokens
constexpr int cNQKV = (cH + 2 * cKV) * cHD; // 3072

#if defined(__has_builtin)
#if __has_builtin(__builtin_amdgcn_global_load_async_to_lds_b128)
#define HAVE_ASYNC 1
#endif
#endif
#ifndef HAVE_ASYNC
#define HAVE_ASYNC 0
#endif

static __device__ __forceinline__ v8f wmma_bf16(v16bf a, v16bf b, v8f c) {
    return __builtin_amdgcn_wmma_f32_16x16x32_bf16(false, a, false, b, (short)0, c,
                                                   false, false);
}

// Fragment from bf16 row (A operand, or B column row in transposed LDS).
// CDNA5 16-bit layout: lanes 0-15 hold K 0..7 / 16..23, lanes 16-31 hold
// K 8..15 / 24..31 -> two 16-byte loads per lane.
static __device__ __forceinline__ v16bf load_frag16(const bf16_t* rowp, int hi) {
    v8bf x = *(const v8bf*)(rowp + hi * 8);
    v8bf y = *(const v8bf*)(rowp + 16 + hi * 8);
    v16bf r;
#pragma unroll
    for (int i = 0; i < 8; i++) { r[i] = x[i]; r[i + 8] = y[i]; }
    return r;
}

// Fragment from fp32 LDS tile in natural [k][128n] layout: 16 strided
// ds_load_b32 + cvt per lane. Consecutive lanes -> consecutive banks.
static __device__ __forceinline__ v16bf load_frag16_f32lds(const float* colp, int hi) {
    v16bf r;
#pragma unroll
    for (int i = 0; i < 8; i++) {
        r[i]     = (bf16_t)colp[(hi * 8 + i) * 128];
        r[i + 8] = (bf16_t)colp[(16 + hi * 8 + i) * 128];
    }
    return r;
}

// Stage a 32k x 128n fp32 tile from global (row stride ldn) into LDS in
// natural layout. Async path: 4 GLOBAL_LOAD_ASYNC_TO_LDS_B128 per thread.
template <bool NGUARD>
static __device__ __forceinline__ void stage_tile_f32(float* dst, const float* src,
                                                      int ldn, int tid, int nb, int N) {
#pragma unroll
    for (int c = 0; c < 4; c++) {
        int linear = tid + c * 256;      // 1024 float4 chunks
        int k = linear >> 5, n4 = linear & 31;
        if constexpr (NGUARD) {
#pragma unroll
            for (int j = 0; j < 4; j++) {
                int gn = nb + n4 * 4 + j;
                dst[k * 128 + n4 * 4 + j] =
                    (gn < N) ? src[(size_t)k * ldn + n4 * 4 + j] : 0.f;
            }
        } else {
#if HAVE_ASYNC
            __builtin_amdgcn_global_load_async_to_lds_b128(
                (as1_v4i32p)(src + (size_t)k * ldn + n4 * 4),
                (as3_v4i32p)(dst + k * 128 + n4 * 4), 0, 0);
#else
            *(float4*)(dst + k * 128 + n4 * 4) =
                *(const float4*)(src + (size_t)k * ldn + n4 * 4);
            __builtin_prefetch(src + (size_t)(k + 32) * ldn + n4 * 4, 0, 1);
#endif
        }
    }
}

// Wait for async staging. pending=true: one tile group (4 instr/thread) may
// remain in flight (double-buffer prefetch); pending=false: drain.
static __device__ __forceinline__ void wait_stage(bool pending) {
#if HAVE_ASYNC
    if (pending) asm volatile("s_wait_asynccnt 0x4" ::: "memory");
    else         asm volatile("s_wait_asynccnt 0x0" ::: "memory");
#endif
}

// ---------------------------------------------------------------------------
// RMSNorm (fp32 in -> bf16 out), one row (D=2048) per 256-thread block.
// ---------------------------------------------------------------------------
__global__ __launch_bounds__(256) void k_rmsnorm(const float* __restrict__ X,
                                                 const float* __restrict__ w,
                                                 bf16_t* __restrict__ Y) {
    int row = blockIdx.x;
    const float* xr = X + (size_t)row * cD;
    bf16_t* yr = Y + (size_t)row * cD;
    float s = 0.f;
    for (int i = threadIdx.x; i < cD; i += 256) { float v = xr[i]; s += v * v; }
#pragma unroll
    for (int off = 16; off; off >>= 1) s += __shfl_xor(s, off);
    __shared__ float red[8];
    int wv = threadIdx.x >> 5, lane = threadIdx.x & 31;
    if (lane == 0) red[wv] = s;
    __syncthreads();
    float tot = red[0] + red[1] + red[2] + red[3] + red[4] + red[5] + red[6] + red[7];
    float inv = rsqrtf(tot / cD + 1e-6f);
    for (int i = threadIdx.x; i < cD; i += 256)
        yr[i] = (bf16_t)(xr[i] * inv * w[i]);
}

// ---------------------------------------------------------------------------
// Generic WMMA GEMM: C[M,N] = A(bf16,row-major) * B(fp32,row-major).
// Block = 256 threads = 8 waves (4M x 2N); tile 64M x 128N, K-step 32.
// Double-buffered async B staging: next tile streams under ASYNCcnt while
// WMMAs consume the current tile. CMODE 0: C = acc; 1: C = bias + acc.
// ---------------------------------------------------------------------------
template <int CMODE, bool NGUARD>
__global__ __launch_bounds__(256) void k_gemm(const bf16_t* __restrict__ A,
                                              const float* __restrict__ Bw,
                                              const float* __restrict__ bias,
                                              float* __restrict__ C,
                                              int M, int N, int Kd) {
    __shared__ alignas(16) float sB[2][32 * 128];
    int tid = threadIdx.x, lane = tid & 31, ml = lane & 15, hi = lane >> 4;
    int wave = tid >> 5, wm = wave & 3, wn = wave >> 2;
    int m0 = blockIdx.y * 64 + wm * 16;
    int nb = blockIdx.x * 128;
    v8f acc[4] = {};
    int rowA = m0 + ml; if (rowA >= M) rowA = M - 1;
    const bf16_t* arow = A + (size_t)rowA * Kd;
    int nsteps = Kd / 32;
    stage_tile_f32<NGUARD>(sB[0], Bw + nb, N, tid, nb, N);
    for (int s = 0; s < nsteps; s++) {
        int k0 = s * 32;
        float* cur = sB[s & 1];
        bool pre = (s + 1 < nsteps);
        if (pre)
            stage_tile_f32<NGUARD>(sB[(s + 1) & 1], Bw + (size_t)(k0 + 32) * N + nb, N,
                                   tid, nb, N);
        wait_stage(pre);
        __syncthreads();
        v16bf af = load_frag16(arow + k0, hi);
#pragma unroll
        for (int f = 0; f < 4; f++) {
            v16bf bfg = load_frag16_f32lds(&cur[wn * 64 + f * 16 + ml], hi);
            acc[f] = wmma_bf16(af, bfg, acc[f]);
        }
        __syncthreads();
    }
#pragma unroll
    for (int f = 0; f < 4; f++) {
        int col = nb + wn * 64 + f * 16 + ml;
        if (col >= N) continue;
#pragma unroll
        for (int r = 0; r < 8; r++) {
            int row = m0 + r + 8 * hi;
            if (row >= M) continue;
            float v = acc[f][r];
            if constexpr (CMODE == 1) v += bias[(size_t)row * N + col];
            C[(size_t)row * N + col] = v;
        }
    }
}

// ---------------------------------------------------------------------------
// QK-norm + RoPE + bf16 repack. One wave per (token, 128-wide head row).
// slots: 0..15 = q heads, 16..19 = k heads, 20..23 = v heads.
// ---------------------------------------------------------------------------
__global__ __launch_bounds__(256) void k_qkv_post(const float* __restrict__ qkv,
                                                  const int* __restrict__ positions,
                                                  const float* __restrict__ qw,
                                                  const float* __restrict__ kw,
                                                  bf16_t* __restrict__ Q,
                                                  bf16_t* __restrict__ Kb,
                                                  bf16_t* __restrict__ Vb) {
    int lane = threadIdx.x & 31;
    int gw = (blockIdx.x * blockDim.x + threadIdx.x) >> 5;
    int slot = gw % 24;
    int t = gw / 24;
    if (t >= cT) return;
    const float* src = qkv + (size_t)t * cNQKV + slot * cHD;
    float x[4];
#pragma unroll
    for (int j = 0; j < 4; j++) x[j] = src[lane * 4 + j];
    if (slot < cH + cKV) {
        float ss = x[0] * x[0] + x[1] * x[1] + x[2] * x[2] + x[3] * x[3];
#pragma unroll
        for (int off = 16; off; off >>= 1) ss += __shfl_xor(ss, off);
        float inv = rsqrtf(ss / cHD + 1e-6f);
        const float* w = (slot < cH) ? qw : kw;
#pragma unroll
        for (int j = 0; j < 4; j++) x[j] = x[j] * inv * w[lane * 4 + j];
        int pos = positions[t];
        int l16 = lane & 15, hi = lane >> 4;
#pragma unroll
        for (int j = 0; j < 4; j++) {
            int i = l16 * 4 + j;  // freq index 0..63
            float freq = (float)pos * __powf(10000.0f, -(float)i / 64.0f);
            float sn, cs; __sincosf(freq, &sn, &cs);
            float other = __shfl_xor(x[j], 16);
            x[j] = hi ? (x[j] * cs + other * sn) : (x[j] * cs - other * sn);
        }
    }
    bf16_t* dst;
    if (slot < cH)             dst = Q  + ((size_t)t * cH  + slot) * cHD;
    else if (slot < cH + cKV)  dst = Kb + ((size_t)t * cKV + (slot - cH)) * cHD;
    else                       dst = Vb + ((size_t)t * cKV + (slot - cH - cKV)) * cHD;
#pragma unroll
    for (int j = 0; j < 4; j++) dst[lane * 4 + j] = (bf16_t)x[j];
}

// ---------------------------------------------------------------------------
// Flash attention: one wave per (b, h, 16-query tile). QK^T via WMMA with K
// rows read directly as B-fragments; all 8 fragments preloaded so the b128
// loads pipeline ahead of the WMMA chain. Online softmax via half-wave
// shuffles. P bounced through wave-private LDS (C->A relayout); V staged
// transposed in wave-private LDS with vectorized b128 global reads.
// ---------------------------------------------------------------------------
__global__ __launch_bounds__(128) void k_attn(const bf16_t* __restrict__ Q,
                                              const bf16_t* __restrict__ Kb,
                                              const bf16_t* __restrict__ Vb,
                                              bf16_t* __restrict__ Oa) {
    __shared__ alignas(16) bf16_t sP[4][16 * 32];
    __shared__ alignas(16) bf16_t sV[4][128 * 40];
    int lane = threadIdx.x & 31, ml = lane & 15, hi = lane >> 4;
    int wv = threadIdx.x >> 5;
    int gw = blockIdx.x * 4 + wv;
    const int QT = cS / 16;
    if (gw >= cB * cH * QT) return;
    int qt = gw % QT, h = (gw / QT) % cH, b = gw / (QT * cH);
    int q0 = qt * 16;
    int kvh = h / (cH / cKV);
    const bf16_t* qrow = Q + ((size_t)(b * cS + q0 + ml) * cH + h) * cHD;
    v16bf aq[4];
#pragma unroll
    for (int c = 0; c < 4; c++) aq[c] = load_frag16(qrow + c * 32, hi);
    float mrow[8], lrow[8];
#pragma unroll
    for (int r = 0; r < 8; r++) { mrow[r] = -3.0e38f; lrow[r] = 0.f; }
    v8f oacc[8] = {};
    const float scale = 0.08838834764831845f;  // 1/sqrt(128)
    bf16_t* pP = &sP[wv][0];
    bf16_t* pV = &sV[wv][0];
    int kbmax = (q0 + 15) / 32;
    for (int kb = 0; kb <= kbmax; kb++) {
        // ---- scores: preload all 8 B fragments, then WMMA chain ----
        const bf16_t* k0r = Kb + ((size_t)(b * cS + kb * 32 + ml) * cKV + kvh) * cHD;
        const bf16_t* k1r = k0r + (size_t)16 * cKV * cHD;
        v16bf bk[8];
#pragma unroll
        for (int c = 0; c < 4; c++) {
            bk[c]     = load_frag16(k0r + c * 32, hi);
            bk[4 + c] = load_frag16(k1r + c * 32, hi);
        }
        v8f s0 = {}, s1 = {};
#pragma unroll
        for (int c = 0; c < 4; c++) {
            s0 = wmma_bf16(aq[c], bk[c], s0);
            s1 = wmma_bf16(aq[c], bk[4 + c], s1);
        }
        // ---- online softmax over the 32 new keys ----
#pragma unroll
        for (int r = 0; r < 8; r++) {
            int qi = q0 + r + 8 * hi;
            int ki0 = kb * 32 + ml, ki1 = ki0 + 16;
            float v0 = (ki0 <= qi) ? s0[r] * scale : -3.0e38f;
            float v1 = (ki1 <= qi) ? s1[r] * scale : -3.0e38f;
            float mx = fmaxf(v0, v1);
#pragma unroll
            for (int off = 8; off; off >>= 1) mx = fmaxf(mx, __shfl_xor(mx, off));
            float mnew = fmaxf(mrow[r], mx);
            float p0 = __expf(v0 - mnew), p1 = __expf(v1 - mnew);
            float rs = p0 + p1;
#pragma unroll
            for (int off = 8; off; off >>= 1) rs += __shfl_xor(rs, off);
            float corr = __expf(mrow[r] - mnew);
            lrow[r] = lrow[r] * corr + rs;
            mrow[r] = mnew;
#pragma unroll
            for (int f = 0; f < 8; f++) oacc[f][r] *= corr;
            pP[(r + 8 * hi) * 32 + ml] = (bf16_t)p0;
            pP[(r + 8 * hi) * 32 + 16 + ml] = (bf16_t)p1;
        }
        // ---- stage V^T [d][k] (stride 40) with vectorized global reads ----
#pragma unroll
        for (int c = 0; c < 16; c++) {
            int linear = lane + c * 32;       // 512 chunks of 8 d-values
            int kk = linear >> 4;
            int db = (linear & 15) * 8;
            v8bf vv = *(const v8bf*)(Vb +
                ((size_t)(b * cS + kb * 32 + kk) * cKV + kvh) * cHD + db);
#pragma unroll
            for (int j = 0; j < 8; j++) pV[(db + j) * 40 + kk] = vv[j];
        }
        // ---- PV WMMAs ----
        v16bf pa = load_frag16(pP + ml * 32, hi);
#pragma unroll
        for (int f = 0; f < 8; f++) {
            v16bf bv = load_frag16(pV + (f * 16 + ml) * 40, hi);
            oacc[f] = wmma_bf16(pa, bv, oacc[f]);
        }
    }
#pragma unroll
    for (int f = 0; f < 8; f++) {
#pragma unroll
        for (int r = 0; r < 8; r++) {
            int qi = q0 + r + 8 * hi;
            float o = oacc[f][r] / lrow[r];
            Oa[((size_t)(b * cS + qi) * cH + h) * cHD + f * 16 + ml] = (bf16_t)o;
        }
    }
}

// ---------------------------------------------------------------------------
// Router: softmax over 64 logits, iterative top-8 via wave argmax reduction.
// ---------------------------------------------------------------------------
__global__ __launch_bounds__(256) void k_route(const float* __restrict__ logits,
                                               int* __restrict__ counts,
                                               int* __restrict__ topi,
                                               float* __restrict__ topw) {
    int lane = threadIdx.x & 31;
    int gw = (blockIdx.x * blockDim.x + threadIdx.x) >> 5;
    if (gw >= cT) return;
    const float* lg = logits + (size_t)gw * cE;
    float v0 = lg[lane], v1 = lg[lane + 32];
    float mx = fmaxf(v0, v1);
#pragma unroll
    for (int off = 16; off; off >>= 1) mx = fmaxf(mx, __shfl_xor(mx, off));
    float e0 = __expf(v0 - mx), e1 = __expf(v1 - mx);
    float s = e0 + e1;
#pragma unroll
    for (int off = 16; off; off >>= 1) s += __shfl_xor(s, off);
    float r0 = e0 / s, r1 = e1 / s;
    float wsum = 0.f;
    float pv[cK]; int pe[cK];
#pragma unroll
    for (int k = 0; k < cK; k++) {
        float bv; int bi;
        if (r0 >= r1) { bv = r0; bi = lane; } else { bv = r1; bi = lane + 32; }
#pragma unroll
        for (int off = 16; off; off >>= 1) {
            float ov = __shfl_xor(bv, off);
            int   oi = __shfl_xor(bi, off);
            if (ov > bv || (ov == bv && oi < bi)) { bv = ov; bi = oi; }
        }
        pv[k] = bv; pe[k] = bi; wsum += bv;
        if (bi == lane)      r0 = -1.f;
        if (bi == lane + 32) r1 = -1.f;
    }
    if (lane == 0) {
#pragma unroll
        for (int k = 0; k < cK; k++) {
            topi[gw * cK + k] = pe[k];
            topw[gw * cK + k] = pv[k] / wsum;
            atomicAdd(&counts[pe[k]], 1);
        }
    }
}

__global__ void k_scan(const int* __restrict__ counts, int* __restrict__ offsets,
                       int* __restrict__ cursor) {
    if (threadIdx.x == 0) {
        int acc = 0;
        for (int e = 0; e < cE; e++) { offsets[e] = acc; cursor[e] = acc; acc += counts[e]; }
    }
}

__global__ __launch_bounds__(256) void k_scatter(const int* __restrict__ topi,
                                                 const float* __restrict__ topw,
                                                 int* __restrict__ cursor,
                                                 int* __restrict__ tok_idx,
                                                 float* __restrict__ tok_w) {
    int idx = blockIdx.x * blockDim.x + threadIdx.x;
    if (idx >= cT * cK) return;
    int e = topi[idx];
    int pos = atomicAdd(&cursor[e], 1);
    tok_idx[pos] = idx / cK;
    tok_w[pos] = topw[idx];
}

// ---------------------------------------------------------------------------
// Expert gate+up GEMM with gathered A rows, fused SiLU(g)*u, bf16 act out.
// Both tile groups stream under ASYNCcnt; single-buffered (2x16KB LDS).
// ---------------------------------------------------------------------------
__global__ __launch_bounds__(256) void k_moe_gateup(const bf16_t* __restrict__ X,
                                                    const float* __restrict__ Wgu,
                                                    const int* __restrict__ counts,
                                                    const int* __restrict__ offsets,
                                                    const int* __restrict__ tok_idx,
                                                    bf16_t* __restrict__ Act) {
    int e = blockIdx.z;
    int cnt = counts[e];
    if ((int)blockIdx.y * 64 >= cnt) return;
    int base = offsets[e];
    __shared__ alignas(16) float sG[32 * 128];
    __shared__ alignas(16) float sU[32 * 128];
    int tid = threadIdx.x, lane = tid & 31, ml = lane & 15, hi = lane >> 4;
    int wave = tid >> 5, wm = wave & 3, wn = wave >> 2;
    int m0 = blockIdx.y * 64 + wm * 16;
    int nb = blockIdx.x * 128;
    const float* W = Wgu + (size_t)e * cD * (2 * cI);
    v8f ag[4] = {}; v8f au[4] = {};
    int m = m0 + ml; if (m >= cnt) m = cnt - 1;
    int tok = tok_idx[base + m];
    const bf16_t* arow = X + (size_t)tok * cD;
    for (int k0 = 0; k0 < cD; k0 += 32) {
        const float* wrow = W + (size_t)k0 * (2 * cI) + nb;
        stage_tile_f32<false>(sG, wrow,       2 * cI, tid, 0, 0);
        stage_tile_f32<false>(sU, wrow + cI,  2 * cI, tid, 0, 0);
        wait_stage(false);
        __syncthreads();
        v16bf af = load_frag16(arow + k0, hi);
#pragma unroll
        for (int f = 0; f < 4; f++) {
            ag[f] = wmma_bf16(af, load_frag16_f32lds(&sG[wn * 64 + f * 16 + ml], hi), ag[f]);
            au[f] = wmma_bf16(af, load_frag16_f32lds(&sU[wn * 64 + f * 16 + ml], hi), au[f]);
        }
        __syncthreads();
    }
#pragma unroll
    for (int f = 0; f < 4; f++) {
        int col = nb + wn * 64 + f * 16 + ml;
#pragma unroll
        for (int r = 0; r < 8; r++) {
            int row = m0 + r + 8 * hi;
            if (row >= cnt) continue;
            float g = ag[f][r], u = au[f][r];
            float a = (g / (1.f + __expf(-g))) * u;
            Act[(size_t)(base + row) * cI + col] = (bf16_t)a;
        }
    }
}

// ---------------------------------------------------------------------------
// Expert down GEMM over compact activation rows; double-buffered async B
// staging; scaled atomic scatter-add (global_atomic_add_f32) into output.
// ---------------------------------------------------------------------------
__global__ __launch_bounds__(256) void k_moe_down(const bf16_t* __restrict__ Act,
                                                  const float* __restrict__ Wd,
                                                  const int* __restrict__ counts,
                                                  const int* __restrict__ offsets,
                                                  const int* __restrict__ tok_idx,
                                                  const float* __restrict__ tok_w,
                                                  float* __restrict__ Out) {
    int e = blockIdx.z;
    int cnt = counts[e];
    if ((int)blockIdx.y * 64 >= cnt) return;
    int base = offsets[e];
    __shared__ alignas(16) float sB[2][32 * 128];
    int tid = threadIdx.x, lane = tid & 31, ml = lane & 15, hi = lane >> 4;
    int wave = tid >> 5, wm = wave & 3, wn = wave >> 2;
    int m0 = blockIdx.y * 64 + wm * 16;
    int nb = blockIdx.x * 128;
    const float* W = Wd + (size_t)e * cI * cD;
    v8f acc[4] = {};
    int m = m0 + ml; if (m >= cnt) m = cnt - 1;
    const bf16_t* arow = Act + (size_t)(base + m) * cI;
    const int nsteps = cI / 32;
    stage_tile_f32<false>(sB[0], W + nb, cD, tid, 0, 0);
    for (int s = 0; s < nsteps; s++) {
        int k0 = s * 32;
        float* cur = sB[s & 1];
        bool pre = (s + 1 < nsteps);
        if (pre)
            stage_tile_f32<false>(sB[(s + 1) & 1], W + (size_t)(k0 + 32) * cD + nb, cD,
                                  tid, 0, 0);
        wait_stage(pre);
        __syncthreads();
        v16bf af = load_frag16(arow + k0, hi);
#pragma unroll
        for (int f = 0; f < 4; f++)
            acc[f] = wmma_bf16(af, load_frag16_f32lds(&cur[wn * 64 + f * 16 + ml], hi),
                               acc[f]);
        __syncthreads();
    }
#pragma unroll
    for (int f = 0; f < 4; f++) {
        int col = nb + wn * 64 + f * 16 + ml;
#pragma unroll
        for (int r = 0; r < 8; r++) {
            int row = m0 + r + 8 * hi;
            if (row >= cnt) continue;
            int tok = tok_idx[base + row];
            float w = tok_w[base + row];
            unsafeAtomicAdd(&Out[(size_t)tok * cD + col], w * acc[f][r]);
        }
    }
}

__global__ void k_zero_f32(float* p, long long n) {
    long long i = (long long)blockIdx.x * blockDim.x + threadIdx.x;
    if (i < n) p[i] = 0.f;
}
__global__ void k_zero_i32(int* p, int n) {
    int i = blockIdx.x * blockDim.x + threadIdx.x;
    if (i < n) p[i] = 0;
}

// ---------------------------------------------------------------------------
extern "C" void kernel_launch(void* const* d_in, const int* in_sizes, int n_in,
                              void* d_out, int out_size, void* d_ws, size_t ws_size,
                              hipStream_t stream) {
    const int*   positions = (const int*)d_in[0];
    const float* hidden    = (const float*)d_in[1];
    const float* ln1_w     = (const float*)d_in[2];
    const float* wqkv      = (const float*)d_in[3];
    const float* q_norm_w  = (const float*)d_in[4];
    const float* k_norm_w  = (const float*)d_in[5];
    const float* wo        = (const float*)d_in[6];
    const float* ln2_w     = (const float*)d_in[7];
    const float* gate_w    = (const float*)d_in[8];
    const float* w_gate_up = (const float*)d_in[9];
    const float* w_down    = (const float*)d_in[10];

    float* out_moe  = (float*)d_out;                      // [T, D]
    float* residual = (float*)d_out + (size_t)cT * cD;    // [T, D]

    char* ws = (char*)d_ws;
    auto alloc = [&](size_t bytes) -> char* {
        char* p = ws;
        ws += (bytes + 255) & ~(size_t)255;
        return p;
    };
    bf16_t* h1      = (bf16_t*)alloc((size_t)cT * cD * 2);
    float*  qkv     = (float*) alloc((size_t)cT * cNQKV * 4);
    bf16_t* qb      = (bf16_t*)alloc((size_t)cT * cH * cHD * 2);
    bf16_t* kb      = (bf16_t*)alloc((size_t)cT * cKV * cHD * 2);
    bf16_t* vb      = (bf16_t*)alloc((size_t)cT * cKV * cHD * 2);
    bf16_t* attn    = (bf16_t*)alloc((size_t)cT * cH * cHD * 2);
    bf16_t* h2      = (bf16_t*)alloc((size_t)cT * cD * 2);
    float*  logits  = (float*) alloc((size_t)cT * cE * 4);
    int*    counts  = (int*)   alloc(cE * 4);
    int*    offsets = (int*)   alloc(cE * 4);
    int*    cursor  = (int*)   alloc(cE * 4);
    int*    topi    = (int*)   alloc((size_t)cT * cK * 4);
    float*  topw    = (float*) alloc((size_t)cT * cK * 4);
    int*    tok_idx = (int*)   alloc((size_t)cT * cK * 4);
    float*  tok_w   = (float*) alloc((size_t)cT * cK * 4);
    bf16_t* act     = (bf16_t*)alloc((size_t)cT * cK * cI * 2);

    dim3 b256(256), b128(128);

    // 1) h1 = rmsnorm(hidden) -> bf16
    k_rmsnorm<<<cT, b256, 0, stream>>>(hidden, ln1_w, h1);
    // 2) qkv = h1 @ wqkv
    k_gemm<0, false><<<dim3(cNQKV / 128, cT / 64), b256, 0, stream>>>(
        h1, wqkv, nullptr, qkv, cT, cNQKV, cD);
    // 3) q/k norm + rope -> bf16 q/k/v
    k_qkv_post<<<(cT * 24) / 8, b256, 0, stream>>>(qkv, positions, q_norm_w, k_norm_w,
                                                   qb, kb, vb);
    // 4) flash attention
    k_attn<<<(cB * cH * (cS / 16)) / 4, b128, 0, stream>>>(qb, kb, vb, attn);
    // 5) residual = hidden + attn @ wo   (fused bias add)
    k_gemm<1, false><<<dim3(cD / 128, cT / 64), b256, 0, stream>>>(
        attn, wo, hidden, residual, cT, cD, cD);
    // 6) h2 = rmsnorm(residual) -> bf16
    k_rmsnorm<<<cT, b256, 0, stream>>>(residual, ln2_w, h2);
    // 7) router logits (N=64 < tile -> guarded manual staging path)
    k_gemm<0, true><<<dim3(1, cT / 64), b256, 0, stream>>>(h2, gate_w, nullptr, logits,
                                                           cT, cE, cD);
    // 8) routing + compact per-expert lists
    k_zero_i32<<<1, 64, 0, stream>>>(counts, cE);
    k_route<<<cT / 8, b256, 0, stream>>>(logits, counts, topi, topw);
    k_scan<<<1, 32, 0, stream>>>(counts, offsets, cursor);
    k_scatter<<<(cT * cK) / 256, b256, 0, stream>>>(topi, topw, cursor, tok_idx, tok_w);
    // 9) MoE: zero out accumulator, gathered gate-up+SiLU, down + scatter-add
    k_zero_f32<<<(unsigned)(((long long)cT * cD + 255) / 256), b256, 0, stream>>>(
        out_moe, (long long)cT * cD);
    k_moe_gateup<<<dim3(cI / 128, cT / 64, cE), b256, 0, stream>>>(h2, w_gate_up, counts,
                                                                   offsets, tok_idx, act);
    k_moe_down<<<dim3(cD / 128, cT / 64, cE), b256, 0, stream>>>(act, w_down, counts,
                                                                 offsets, tok_idx, tok_w,
                                                                 out_moe);
}
